// SequenceGenerator_17051020165413
// MI455X (gfx1250) — matrix operand order repaired and checked by reading, not compiled, not run on hardware
//
#include <hip/hip_runtime.h>
#include <hip/hip_bf16.h>

// ---------------------------------------------------------------------------
// Autoregressive LSTM + Gumbel-hard sampling, CDNA5 (gfx1250) wave32 WMMA.
// B=2048, T=256, H=1024, C=128.  Dominant work: per-step [2048x1024]x[1024x4096]
// bf16 WMMA GEMM with f32 accumulate.  x is one-hot -> W_ih contribution is a
// gather; sampling is argmax(logits + gumbel).  K-loop is double-buffered
// (unroll 2, peeled tail): loads land directly in the registers their WMMA
// consumes, no rotation moves.  One M-tile x 4 gate accumulators per wave.
// ---------------------------------------------------------------------------

#define Bsz 2048
#define Tt  256
#define Hh  1024
#define Cc  128

typedef __attribute__((ext_vector_type(16))) __bf16 v16bf;
typedef __attribute__((ext_vector_type(8)))  __bf16 v8bf;
typedef __attribute__((ext_vector_type(8)))  float  v8f;

union AFrag { v16bf v; v8bf h8[2]; };

__device__ __forceinline__ float sigf(float x) { return 1.0f / (1.0f + __expf(-x)); }

// Load one A fragment (16x32 of h, bf16) + four gate B fragments for k-tile kt.
__device__ __forceinline__ void load_frags4(const __bf16* __restrict__ arow,
                                            const __bf16* const* bptr,
                                            int kt, int half,
                                            AFrag& A, v16bf B[4]) {
    const int kb = kt * 32;
    A.h8[0] = *(const v8bf*)(arow + kb + half * 8);
    A.h8[1] = *(const v8bf*)(arow + kb + 16 + half * 8);
#pragma unroll
    for (int g = 0; g < 4; ++g)
        B[g] = *(const v16bf*)(bptr[g] + (size_t)kt * 512);
}

__device__ __forceinline__ void mma4(const AFrag& A, const v16bf B[4], v8f acc[4]) {
#pragma unroll
    for (int g = 0; g < 4; ++g)
        acc[g] = __builtin_amdgcn_wmma_f32_16x16x32_bf16(
            false, A.v, false, B[g], (short)0, acc[g], false, false);
}

// ---------------------------------------------------------------------------
// Pack a row-major f32 weight matrix W[Nrows x K] into bf16 WMMA B-fragment
// tiles for D = A(16x32) * B(32x16):  B[k][n] = W[n][k].
// Tile (ntile, kt) is 512 bf16: lane L owns 16 contiguous values,
//   n = ntile*16 + (L&15),  k = kt*32 + (j>>3)*16 + (L>>4)*8 + (j&7)  (j = 0..15).
// ---------------------------------------------------------------------------
__global__ void pack_b_kernel(const float* __restrict__ W, __bf16* __restrict__ dst,
                              int Nrows, int K) {
    const int KT = K / 32;
    const size_t total = (size_t)Nrows * (size_t)K;
    for (size_t gid = (size_t)blockIdx.x * blockDim.x + threadIdx.x; gid < total;
         gid += (size_t)gridDim.x * blockDim.x) {
        int blk    = (int)(gid >> 9);       // /512
        int within = (int)(gid & 511);
        int lane   = within >> 4;
        int j      = within & 15;
        int ntile  = blk / KT;
        int kt     = blk % KT;
        int n      = ntile * 16 + (lane & 15);
        int k      = kt * 32 + (j >> 3) * 16 + (lane >> 4) * 8 + (j & 7);
        dst[gid] = (__bf16)W[(size_t)n * K + k];
    }
}

// ---------------------------------------------------------------------------
// One-shot prep: W_ih transpose -> [C,4H] f32 (coalesced gather), bias fuse,
// c=0, h0 = bf16(noise), x_idx = -1 (x0 = zero vector).
// ---------------------------------------------------------------------------
__global__ void prep_misc_kernel(const float* __restrict__ W_ih,
                                 const float* __restrict__ b_ih,
                                 const float* __restrict__ b_hh,
                                 const float* __restrict__ noise,
                                 float* __restrict__ wihT,
                                 float* __restrict__ bias,
                                 float* __restrict__ c_st,
                                 __bf16* __restrict__ h0,
                                 int* __restrict__ xidx) {
    const size_t total = (size_t)Bsz * Hh;   // largest range: 2,097,152
    for (size_t gid = (size_t)blockIdx.x * blockDim.x + threadIdx.x; gid < total;
         gid += (size_t)gridDim.x * blockDim.x) {
        c_st[gid] = 0.0f;
        h0[gid]   = (__bf16)noise[gid];
        if (gid < (size_t)Cc * 4 * Hh) {     // 524,288: W_ih^T
            int c = (int)(gid / (4 * Hh));
            int n = (int)(gid % (4 * Hh));
            wihT[gid] = W_ih[(size_t)n * Cc + c];
        }
        if (gid < 4 * Hh) bias[gid] = b_ih[gid] + b_hh[gid];
        if (gid < Bsz)    xidx[gid] = -1;
    }
}

// ---------------------------------------------------------------------------
// Fused recurrent GEMM + LSTM cell.  Grid: (H/32, B/64), 256 threads = 8 waves
// arranged 4(M) x 2(N).  Each wave owns one 16x16 tile per gate (i,f,g,o) and
// applies the cell update in-register after the K loop.
// A (16x32 bf16) layout: lane L -> row M = L&15; K chunks (L>>4)*8 + {0..7}
// and 16 + (L>>4)*8 + {0..7}  => two aligned b128 loads from h (bf16).
// ---------------------------------------------------------------------------
__global__ __launch_bounds__(256)
void lstm_step_kernel(const __bf16* __restrict__ h_in,
                      __bf16* __restrict__ h_out,
                      float* __restrict__ c_st,
                      const __bf16* __restrict__ whh_pk,
                      const float* __restrict__ wihT,
                      const float* __restrict__ bias,
                      const int* __restrict__ xidx) {
    const int lane  = threadIdx.x & 31;
    const int w     = threadIdx.x >> 5;
    const int wm    = w >> 1;
    const int wn    = w & 1;
    const int mbase = blockIdx.y * 64 + wm * 16;
    const int nbase = blockIdx.x * 32 + wn * 16;   // column within one gate group
    const int nl    = lane & 15;
    const int half  = lane >> 4;
    const int KT    = Hh / 32;

    v8f acc[4] = {};   // i, f, g, o accumulators (f32)

    const int m = mbase + nl;
    const __bf16* arow = h_in + (size_t)m * Hh;

    // Per-gate B-fragment base pointers (lane-resolved, advance by 512/k-tile).
    const __bf16* bptr[4];
#pragma unroll
    for (int g = 0; g < 4; ++g) {
        const int ntile = (g * Hh + nbase) >> 4;
        bptr[g] = whh_pk + (size_t)ntile * KT * 512 + lane * 16;
    }

    // --- double-buffered K loop (unroll 2, peeled tail) --------------------
    AFrag A0, A1;
    v16bf B0[4], B1[4];
    load_frags4(arow, bptr, 0, half, A0, B0);

    for (int kt = 0; kt < KT - 2; kt += 2) {
        load_frags4(arow, bptr, kt + 1, half, A1, B1);
        mma4(A0, B0, acc);
        load_frags4(arow, bptr, kt + 2, half, A0, B0);
        mma4(A1, B1, acc);
    }
    load_frags4(arow, bptr, KT - 1, half, A1, B1);
    mma4(A0, B0, acc);
    mma4(A1, B1, acc);

    // Epilogue: bias + one-hot-x gather + LSTM cell (PyTorch gate order i,f,g,o).
    const int jcol = nbase + nl;
    const float bI = bias[0 * Hh + jcol];
    const float bF = bias[1 * Hh + jcol];
    const float bG = bias[2 * Hh + jcol];
    const float bO = bias[3 * Hh + jcol];

#pragma unroll
    for (int r = 0; r < 8; ++r) {
        const int mr  = mbase + half * 8 + r;     // C/D layout: VGPR r, half-select
        const int idx = xidx[mr];
        float xi = 0.f, xf = 0.f, xg = 0.f, xo = 0.f;
        if (idx >= 0) {
            const float* wr = wihT + (size_t)idx * (4 * Hh);
            xi = wr[0 * Hh + jcol];
            xf = wr[1 * Hh + jcol];
            xg = wr[2 * Hh + jcol];
            xo = wr[3 * Hh + jcol];
        }
        const float gi = acc[0][r] + bI + xi;
        const float gf = acc[1][r] + bF + xf;
        const float gg = acc[2][r] + bG + xg;
        const float go = acc[3][r] + bO + xo;

        const size_t off = (size_t)mr * Hh + jcol;
        const float cn = sigf(gf) * c_st[off] + sigf(gi) * tanhf(gg);
        c_st[off]  = cn;
        h_out[off] = (__bf16)(sigf(go) * tanhf(cn));
    }
}

// ---------------------------------------------------------------------------
// Classifier + Gumbel-hard sampling.  One block per 16 batch rows; 8 waves map
// to the 8 column tiles of C=128.  Writes logits to out[B,T,C] and argmax of
// (logits + gumbel) to xidx (tau > 0 does not change the argmax).
// ---------------------------------------------------------------------------
__global__ __launch_bounds__(256)
void cls_step_kernel(const __bf16* __restrict__ h_in,
                     const __bf16* __restrict__ wcls_pk,
                     const float* __restrict__ b_cls,
                     const float* __restrict__ u,
                     float* __restrict__ out,
                     int* __restrict__ xidx,
                     int t) {
    __shared__ float sval[16 * Cc];

    const int lane  = threadIdx.x & 31;
    const int w     = threadIdx.x >> 5;    // column tile 0..7
    const int mbase = blockIdx.x * 16;
    const int nl    = lane & 15;
    const int half  = lane >> 4;
    const int KT    = Hh / 32;

    v8f acc = {};
    const int m = mbase + nl;
    const __bf16* arow  = h_in + (size_t)m * Hh;
    const __bf16* bbase = wcls_pk + (size_t)w * KT * 512 + lane * 16;

    // Double-buffered K loop (unroll 2, peeled tail).
    AFrag A0, A1;
    v16bf Bf0, Bf1;
    A0.h8[0] = *(const v8bf*)(arow + half * 8);
    A0.h8[1] = *(const v8bf*)(arow + 16 + half * 8);
    Bf0 = *(const v16bf*)(bbase);

    for (int kt = 0; kt < KT - 2; kt += 2) {
        const int k1 = (kt + 1) * 32, k2 = (kt + 2) * 32;
        A1.h8[0] = *(const v8bf*)(arow + k1 + half * 8);
        A1.h8[1] = *(const v8bf*)(arow + k1 + 16 + half * 8);
        Bf1 = *(const v16bf*)(bbase + (size_t)(kt + 1) * 512);
        acc = __builtin_amdgcn_wmma_f32_16x16x32_bf16(
            false, A0.v, false, Bf0, (short)0, acc, false, false);

        A0.h8[0] = *(const v8bf*)(arow + k2 + half * 8);
        A0.h8[1] = *(const v8bf*)(arow + k2 + 16 + half * 8);
        Bf0 = *(const v16bf*)(bbase + (size_t)(kt + 2) * 512);
        acc = __builtin_amdgcn_wmma_f32_16x16x32_bf16(
            false, A1.v, false, Bf1, (short)0, acc, false, false);
    }
    {
        const int k1 = (KT - 1) * 32;
        A1.h8[0] = *(const v8bf*)(arow + k1 + half * 8);
        A1.h8[1] = *(const v8bf*)(arow + k1 + 16 + half * 8);
        Bf1 = *(const v16bf*)(bbase + (size_t)(KT - 1) * 512);
        acc = __builtin_amdgcn_wmma_f32_16x16x32_bf16(
            false, A0.v, false, Bf0, (short)0, acc, false, false);
        acc = __builtin_amdgcn_wmma_f32_16x16x32_bf16(
            false, A1.v, false, Bf1, (short)0, acc, false, false);
    }

    const int ccol = w * 16 + nl;
    const float bc = b_cls[ccol];
#pragma unroll
    for (int r = 0; r < 8; ++r) {
        const int mr    = mbase + half * 8 + r;
        const float lg  = acc[r] + bc;
        out[((size_t)mr * Tt + t) * Cc + ccol] = lg;
        const float uu  = u[((size_t)t * Bsz + mr) * Cc + ccol];
        const float gum = -__logf(-__logf(uu));
        sval[(half * 8 + r) * Cc + ccol] = lg + gum;
    }
    __syncthreads();

    if (threadIdx.x < 16) {
        const float* row = sval + threadIdx.x * Cc;
        float best = row[0];
        int bi = 0;
        for (int c2 = 1; c2 < Cc; ++c2) {
            const float v = row[c2];
            if (v > best) { best = v; bi = c2; }
        }
        xidx[mbase + threadIdx.x] = bi;
    }
}

// ---------------------------------------------------------------------------
extern "C" void kernel_launch(void* const* d_in, const int* in_sizes, int n_in,
                              void* d_out, int out_size, void* d_ws, size_t ws_size,
                              hipStream_t stream) {
    const float* noise = (const float*)d_in[0];   // [B,H]
    const float* u     = (const float*)d_in[1];   // [T,B,C]
    const float* W_ih  = (const float*)d_in[2];   // [4H,C]
    const float* W_hh  = (const float*)d_in[3];   // [4H,H]
    const float* b_ih  = (const float*)d_in[4];   // [4H]
    const float* b_hh  = (const float*)d_in[5];   // [4H]
    const float* W_cls = (const float*)d_in[6];   // [C,H]
    const float* b_cls = (const float*)d_in[7];   // [C]
    float* out = (float*)d_out;                   // [B,T,C]

    // Workspace carve-up (256B-aligned slabs, ~27 MB total).
    char* p = (char*)d_ws;
    auto take = [&p](size_t bytes) {
        char* r = p;
        p += (bytes + 255) & ~(size_t)255;
        return r;
    };
    __bf16* whh_pk  = (__bf16*)take((size_t)4 * Hh * Hh * sizeof(__bf16));   // 8 MB
    __bf16* wcls_pk = (__bf16*)take((size_t)Cc * Hh * sizeof(__bf16));       // 256 KB
    float*  wihT    = (float*) take((size_t)Cc * 4 * Hh * sizeof(float));    // 2 MB
    float*  bias    = (float*) take((size_t)4 * Hh * sizeof(float));         // 16 KB
    float*  c_st    = (float*) take((size_t)Bsz * Hh * sizeof(float));       // 8 MB
    __bf16* hbuf0   = (__bf16*)take((size_t)Bsz * Hh * sizeof(__bf16));      // 4 MB
    __bf16* hbuf1   = (__bf16*)take((size_t)Bsz * Hh * sizeof(__bf16));      // 4 MB
    int*    xidx    = (int*)   take((size_t)Bsz * sizeof(int));              // 8 KB

    // One-shot weight packing / state init (re-run every call: deterministic).
    pack_b_kernel<<<4096, 256, 0, stream>>>(W_hh, whh_pk, 4 * Hh, Hh);
    pack_b_kernel<<<512,  256, 0, stream>>>(W_cls, wcls_pk, Cc, Hh);
    prep_misc_kernel<<<8192, 256, 0, stream>>>(W_ih, b_ih, b_hh, noise,
                                               wihT, bias, c_st, hbuf0, xidx);

    __bf16* hb[2] = {hbuf0, hbuf1};
    for (int t = 0; t < Tt; ++t) {
        const __bf16* hin = hb[t & 1];
        __bf16* hout      = hb[(t + 1) & 1];
        lstm_step_kernel<<<dim3(Hh / 32, Bsz / 64), 256, 0, stream>>>(
            hin, hout, c_st, whh_pk, wihT, bias, xidx);
        cls_step_kernel<<<Bsz / 16, 256, 0, stream>>>(
            hout, wcls_pk, b_cls, u, out, xidx, t);
    }
}
